// MultiheadAttention_19722489823948
// MI455X (gfx1250) — compile-verified, matrix operand order checked by reading, not run on hardware
//
#include <hip/hip_runtime.h>
#include <hip/hip_bf16.h>
#include <math.h>
#include <stdint.h>

// ---------------------------------------------------------------------------
// MultiheadAttention forward for MI455X (gfx1250), wave32 + WMMA bf16.
// B=4, S=2048, E=1024, H=16, D=64.
// ---------------------------------------------------------------------------

typedef __attribute__((ext_vector_type(16))) __bf16 v16bf;
typedef __attribute__((ext_vector_type(8)))  __bf16 v8bf;
typedef __attribute__((ext_vector_type(8)))  float  v8f;

#define MHA_B  4
#define MHA_S  2048
#define MHA_E  1024
#define MHA_H  16
#define MHA_D  64
#define MHA_M  (MHA_B * MHA_S)   // 8192 tokens
#define MHA_N3 (3 * MHA_E)       // 3072

// CDNA5 async global->LDS path (guarded: falls back to load + ds_store).
#if defined(__has_builtin)
# if __has_builtin(__builtin_amdgcn_global_load_async_to_lds_b128) && \
     __has_builtin(__builtin_amdgcn_s_wait_asynccnt)
#  define USE_ASYNC_LDS 1
# endif
#endif
#ifndef USE_ASYNC_LDS
# define USE_ASYNC_LDS 0
#endif

#if USE_ASYNC_LDS
// Builtin signature: (global v4i* src, lds v4i* dst, imm offset, imm cpol).
// Build both pointers via integer casts (flat LDS pointers carry the LDS
// offset in their low 32 bits per the ISA flat->LDS truncation rule).
typedef int b128vec __attribute__((vector_size(16)));
typedef __attribute__((address_space(1))) b128vec* g_b128_p;
typedef __attribute__((address_space(3))) b128vec* l_b128_p;
# define ASYNC_WAIT(n) __builtin_amdgcn_s_wait_asynccnt(n)
#else
# define ASYNC_WAIT(n)
#endif

__device__ __forceinline__ v8f wmma_bf16(v16bf a, v16bf b, v8f c) {
    // 8 args: (neg_a, A, neg_b, B, c_mod, C, reuse_a, reuse_b)
    return __builtin_amdgcn_wmma_f32_16x16x32_bf16(false, a, false, b, (short)0, c,
                                                   false, false);
}

// Load one 16x32 (MxK) bf16 operand tile in the WMMA per-lane register layout.
// ISA layout (16-bit A 16x32): lanes 0-15 hold row M=lane with K {0..7,16..23},
// lanes 16-31 hold row M=lane-16 with K {8..15,24..31}. The B operand mirrors
// this on B^T (N x K), so the same loader serves both when "B" is stored
// N-major with K contiguous.
__device__ __forceinline__ v16bf load16(const __bf16* p) {
    union { v16bf v; v8bf h[2]; } u;
    u.h[0] = *(const v8bf*)(p);        // K + 0..7  (or 8..15 in high half)
    u.h[1] = *(const v8bf*)(p + 16);   // K + 16..23 (or 24..31)
    return u.v;
}

__device__ __forceinline__ v16bf load_tile(const __bf16* base, int ld, int row0, int k0) {
    const int lane = threadIdx.x & 31;
    const int r = lane & 15;
    const int g = (lane >> 4) << 3;    // 0 or 8
    return load16(base + (size_t)(row0 + r) * ld + k0 + g);
}

// Copy 16 bytes global -> LDS (async DMA path when available).
__device__ __forceinline__ void stage16(const __bf16* gsrc, __bf16* ldst) {
#if USE_ASYNC_LDS
    __builtin_amdgcn_global_load_async_to_lds_b128(
        (g_b128_p)(uintptr_t)gsrc,
        (l_b128_p)(uint32_t)(uintptr_t)ldst,
        0, 0);
#else
    *(v8bf*)ldst = *(const v8bf*)gsrc;
#endif
}

// ---------------------------------------------------------------------------
// Kernel 1: fp32 -> bf16 convert, optionally transposing (weights -> N x K)
// ---------------------------------------------------------------------------
__global__ void cvt_bf16_kernel(const float* __restrict__ src, __bf16* __restrict__ dst,
                                int rows, int cols, int transpose) {
    long long i = (long long)blockIdx.x * blockDim.x + threadIdx.x;
    long long n = (long long)rows * cols;
    if (i >= n) return;
    int r = (int)(i / cols);
    int c = (int)(i % cols);
    float v = src[i];
    if (transpose) dst[(size_t)c * rows + r] = (__bf16)v;
    else           dst[i] = (__bf16)v;
}

// ---------------------------------------------------------------------------
// Kernel 2: fused QKV projection. C[8192,3072] = xb * Wqkv + bias, routed
// into Q[b,h,s,d], K[b,h,s,d], and V transposed as Vt[b,h,d,s].
// Block = 128 threads = 4 waves; wave computes a 64x32 tile (4x2 WMMA accs:
// 6 operand loads feed 8 WMMAs). k-loop unrolled x2 for pipelining depth.
// ---------------------------------------------------------------------------
__global__ __launch_bounds__(128) void qkv_gemm_kernel(
    const __bf16* __restrict__ A,    // [8192,1024] tokens x in
    const __bf16* __restrict__ Bt,   // [3072,1024] = W_qkv^T (N x K)
    const float*  __restrict__ bias, // [3072]
    __bf16* __restrict__ Qo,         // [B,H,S,D]
    __bf16* __restrict__ Ko,         // [B,H,S,D]
    __bf16* __restrict__ Vt)         // [B,H,D,S]
{
    const int K = MHA_E;
    const int wave = threadIdx.x >> 5;
    const int lane = threadIdx.x & 31;
    const int g = lane >> 4, c = lane & 15;

    const int nb = MHA_N3 / 64;                    // 48 column blocks
    const int m0 = (blockIdx.x / nb) * 128 + (wave >> 1) * 64;
    const int n0 = (blockIdx.x % nb) * 64 + (wave & 1) * 32;

    v8f acc[4][2] = {};
#pragma unroll 2
    for (int k0 = 0; k0 < K; k0 += 32) {
        v16bf a[4], b[2];
        for (int i = 0; i < 4; ++i) a[i] = load_tile(A, K, m0 + 16 * i, k0);
        for (int i = 0; i < 2; ++i) b[i] = load_tile(Bt, K, n0 + 16 * i, k0);
        for (int mt = 0; mt < 4; ++mt)
            for (int nt = 0; nt < 2; ++nt)
                acc[mt][nt] = wmma_bf16(a[mt], b[nt], acc[mt][nt]);
    }

    // Epilogue: C layout -> row = j + 8*g within tile, col = lane&15.
    for (int mt = 0; mt < 4; ++mt)
        for (int nt = 0; nt < 2; ++nt)
            for (int j = 0; j < 8; ++j) {
                int row = m0 + mt * 16 + j + g * 8;       // token index
                int col = n0 + nt * 16 + c;               // fused-qkv column
                float v = acc[mt][nt][j] + bias[col];
                __bf16 bv = (__bf16)v;
                int b = row >> 11;                        // S = 2048
                int s = row & (MHA_S - 1);
                int h = col / (3 * MHA_D);
                int r = col - h * (3 * MHA_D);
                size_t bh = (size_t)(b * MHA_H + h);
                if (r < MHA_D)
                    Qo[(bh * MHA_S + s) * MHA_D + r] = bv;
                else if (r < 2 * MHA_D)
                    Ko[(bh * MHA_S + s) * MHA_D + (r - MHA_D)] = bv;
                else
                    Vt[(bh * MHA_D + (r - 2 * MHA_D)) * MHA_S + s] = bv;
            }
}

// ---------------------------------------------------------------------------
// Kernel 3: flash attention per (b,h). 8 waves / 256 threads per block; the
// block owns 128 query rows; each 32-key K/V block is staged into LDS ONCE
// per block via async global->LDS DMA, double-buffered so the DMA of block
// i+1 overlaps the WMMA compute of block i (async ops complete in order per
// wave, so s_wait_asynccnt<=2 releases the current block while the next two
// copies stay in flight).
// ---------------------------------------------------------------------------
#define ATT_KP 72   // padded row stride (elements) for K tile rows (64 dims)
#define ATT_VP 40   // padded row stride (elements) for V tile rows (32 keys)

__global__ __launch_bounds__(256) void attn_kernel(
    const __bf16* __restrict__ Q,   // [B,H,S,D]
    const __bf16* __restrict__ Kk,  // [B,H,S,D]
    const __bf16* __restrict__ Vt,  // [B,H,D,S]
    __bf16* __restrict__ Aout)      // [B,S,H*D] bf16 for the output GEMM
{
    __shared__ __align__(16) __bf16 Kbuf[2][32 * ATT_KP]; // 32 keys x 64 dims
    __shared__ __align__(16) __bf16 Vbuf[2][64 * ATT_VP]; // 64 dims x 32 keys
    __shared__ __align__(16) __bf16 Pbuf[8][16 * 32];     // per-wave P tile

    const int tid  = threadIdx.x;
    const int wave = tid >> 5;
    const int lane = tid & 31;
    const int g = lane >> 4, c = lane & 15;

    const int qblocks = MHA_S / 128;                // 16 blocks per head
    const int bh = blockIdx.x / qblocks;            // 0..63
    const int q0 = (blockIdx.x % qblocks) * 128 + wave * 16;

    const __bf16* Qb = Q  + (size_t)bh * MHA_S * MHA_D;
    const __bf16* Kb = Kk + (size_t)bh * MHA_S * MHA_D;
    const __bf16* Vb = Vt + (size_t)bh * MHA_D * MHA_S;

    // Q tile (16x64) as two A-operands (K-steps of 32), resident all loop.
    v16bf qa0 = load_tile(Qb, MHA_D, q0, 0);
    v16bf qa1 = load_tile(Qb, MHA_D, q0, 32);

    v8f o[4] = {};
    float m[8], l[8];
    for (int j = 0; j < 8; ++j) { m[j] = -1e30f; l[j] = 0.0f; }
    const float scale = 0.125f;                     // 1/sqrt(64)

    __bf16* P = &Pbuf[wave][0];

    // Cooperative staging indices: 256 threads move 16B each per matrix.
    const int krow = tid >> 3, kseg = tid & 7;      // K: 32 rows x 8 segs
    const int vdim = tid >> 2, vseg = tid & 3;      // V: 64 rows x 4 segs

    // Prologue: stage key-block 0 into buffer 0 (2 async ops per thread).
    stage16(Kb + (size_t)krow * MHA_D + kseg * 8, &Kbuf[0][krow * ATT_KP + kseg * 8]);
    stage16(Vb + (size_t)vdim * MHA_S + vseg * 8, &Vbuf[0][vdim * ATT_VP + vseg * 8]);

    for (int kb = 0; kb < MHA_S; kb += 32) {
        const int cur = (kb >> 5) & 1;
        // Issue next block's DMA into the other buffer (safe: the previous
        // iteration's trailing barrier means no wave still reads it), then
        // release the current block. Immediate-only wait -> literal 2 / 0.
        if (kb + 32 < MHA_S) {
            const int nxt = cur ^ 1, kn = kb + 32;
            stage16(Kb + (size_t)(kn + krow) * MHA_D + kseg * 8,
                    &Kbuf[nxt][krow * ATT_KP + kseg * 8]);
            stage16(Vb + (size_t)vdim * MHA_S + kn + vseg * 8,
                    &Vbuf[nxt][vdim * ATT_VP + vseg * 8]);
            ASYNC_WAIT(2);
        } else {
            ASYNC_WAIT(0);
        }
        __syncthreads();   // all waves' current-block chunks visible

        // ---- logits: 16 queries x 32 keys = two 16x16 f32 tiles ----
        v8f s0 = {}, s1 = {};
        s0 = wmma_bf16(qa0, load_tile(Kbuf[cur], ATT_KP, 0,  0),  s0);
        s0 = wmma_bf16(qa1, load_tile(Kbuf[cur], ATT_KP, 0,  32), s0);
        s1 = wmma_bf16(qa0, load_tile(Kbuf[cur], ATT_KP, 16, 0),  s1);
        s1 = wmma_bf16(qa1, load_tile(Kbuf[cur], ATT_KP, 16, 32), s1);

        float alpha[8];
        for (int j = 0; j < 8; ++j) {
            float a0 = s0[j] * scale;
            float a1 = s1[j] * scale;
            float mx = fmaxf(a0, a1);
            mx = fmaxf(mx, __shfl_xor(mx, 1, 32));
            mx = fmaxf(mx, __shfl_xor(mx, 2, 32));
            mx = fmaxf(mx, __shfl_xor(mx, 4, 32));
            mx = fmaxf(mx, __shfl_xor(mx, 8, 32));
            float mn = fmaxf(m[j], mx);
            alpha[j] = __expf(m[j] - mn);
            float p0 = __expf(a0 - mn);
            float p1 = __expf(a1 - mn);
            float rs = p0 + p1;
            rs += __shfl_xor(rs, 1, 32);
            rs += __shfl_xor(rs, 2, 32);
            rs += __shfl_xor(rs, 4, 32);
            rs += __shfl_xor(rs, 8, 32);
            l[j] = l[j] * alpha[j] + rs;
            m[j] = mn;
            // stage P (C layout): row = j + 8*g, key cols c and c+16
            int row = j + g * 8;
            P[row * 32 + c]      = (__bf16)p0;
            P[row * 32 + 16 + c] = (__bf16)p1;
        }
        for (int t = 0; t < 4; ++t)
            for (int j = 0; j < 8; ++j) o[t][j] *= alpha[j];

        // Reload P in A layout (same wave: LDS pipe is in-order).
        v16bf pa = load_tile(P, 32, 0, 0);

        // P(16x32) x V(32x64), V dims in LDS rows -> contiguous key loads.
        for (int t = 0; t < 4; ++t)
            o[t] = wmma_bf16(pa, load_tile(Vbuf[cur], ATT_VP, t * 16, 0), o[t]);

        __syncthreads();   // all reads of buffers done before next DMA write
    }

    // Normalize and emit bf16 rows for the output projection: [token, H*D]
    const int b = bh >> 4, h = bh & (MHA_H - 1);
    for (int t = 0; t < 4; ++t)
        for (int j = 0; j < 8; ++j) {
            int qrow = q0 + j + g * 8;
            int d = t * 16 + c;
            float v = o[t][j] / l[j];
            Aout[((size_t)(b * MHA_S + qrow)) * MHA_E + h * MHA_D + d] = (__bf16)v;
        }
}

// ---------------------------------------------------------------------------
// Kernel 4: output projection. out[8192,1024] = attn * W_o + b_o (fp32 out).
// Wave computes 64x32 (4x2 WMMA tiles); k-loop unrolled x2.
// ---------------------------------------------------------------------------
__global__ __launch_bounds__(128) void out_gemm_kernel(
    const __bf16* __restrict__ A,    // [8192,1024]
    const __bf16* __restrict__ Bt,   // [1024,1024] = W_o^T (N x K)
    const float*  __restrict__ bias, // [1024]
    float* __restrict__ out)         // [8192,1024]
{
    const int K = MHA_E, N = MHA_E;
    const int wave = threadIdx.x >> 5;
    const int lane = threadIdx.x & 31;
    const int g = lane >> 4, c = lane & 15;

    const int nb = N / 64;                         // 16 column blocks
    const int m0 = (blockIdx.x / nb) * 128 + (wave >> 1) * 64;
    const int n0 = (blockIdx.x % nb) * 64 + (wave & 1) * 32;

    v8f acc[4][2] = {};
#pragma unroll 2
    for (int k0 = 0; k0 < K; k0 += 32) {
        v16bf a[4], b[2];
        for (int i = 0; i < 4; ++i) a[i] = load_tile(A, K, m0 + 16 * i, k0);
        for (int i = 0; i < 2; ++i) b[i] = load_tile(Bt, K, n0 + 16 * i, k0);
        for (int mt = 0; mt < 4; ++mt)
            for (int nt = 0; nt < 2; ++nt)
                acc[mt][nt] = wmma_bf16(a[mt], b[nt], acc[mt][nt]);
    }

    for (int mt = 0; mt < 4; ++mt)
        for (int nt = 0; nt < 2; ++nt)
            for (int j = 0; j < 8; ++j) {
                int row = m0 + mt * 16 + j + g * 8;
                int col = n0 + nt * 16 + c;
                out[(size_t)row * N + col] = acc[mt][nt][j] + bias[col];
            }
}

// ---------------------------------------------------------------------------
// Host-side launcher (graph-capture safe: launches only).
// ---------------------------------------------------------------------------
extern "C" void kernel_launch(void* const* d_in, const int* in_sizes, int n_in,
                              void* d_out, int out_size, void* d_ws, size_t ws_size,
                              hipStream_t stream) {
    (void)in_sizes; (void)n_in; (void)out_size; (void)ws_size;

    const float* x      = (const float*)d_in[0];   // [4,2048,1024]
    const float* W_qkv  = (const float*)d_in[1];   // [1024,3072]
    const float* b_qkv  = (const float*)d_in[2];   // [3072]
    const float* W_o    = (const float*)d_in[3];   // [1024,1024]
    const float* b_o    = (const float*)d_in[4];   // [1024]
    float*       out    = (float*)d_out;           // [4,2048,1024]

    char* ws = (char*)d_ws;
    size_t off = 0;
    __bf16* xb    = (__bf16*)(ws + off); off += (size_t)MHA_M  * MHA_E  * 2; // 16 MB
    __bf16* wqkvT = (__bf16*)(ws + off); off += (size_t)MHA_N3 * MHA_E  * 2; //  6 MB
    __bf16* woT   = (__bf16*)(ws + off); off += (size_t)MHA_E  * MHA_E  * 2; //  2 MB
    __bf16* Qb    = (__bf16*)(ws + off); off += (size_t)MHA_M  * MHA_E  * 2; // 16 MB
    __bf16* Kb    = (__bf16*)(ws + off); off += (size_t)MHA_M  * MHA_E  * 2; // 16 MB
    __bf16* Vt    = (__bf16*)(ws + off); off += (size_t)MHA_M  * MHA_E  * 2; // 16 MB
    __bf16* attnb = (__bf16*)(ws + off); off += (size_t)MHA_M  * MHA_E  * 2; // 16 MB

    // 1) convert / transpose weights and activations to bf16
    {
        int n = MHA_M * MHA_E;
        cvt_bf16_kernel<<<(n + 255) / 256, 256, 0, stream>>>(x, xb, MHA_M, MHA_E, 0);
        n = MHA_E * MHA_N3;
        cvt_bf16_kernel<<<(n + 255) / 256, 256, 0, stream>>>(W_qkv, wqkvT, MHA_E, MHA_N3, 1);
        n = MHA_E * MHA_E;
        cvt_bf16_kernel<<<(n + 255) / 256, 256, 0, stream>>>(W_o, woT, MHA_E, MHA_E, 1);
    }

    // 2) fused QKV projection (WMMA), routes Q/K and transposed V
    {
        dim3 grid((MHA_M / 128) * (MHA_N3 / 64));  // 64 * 48 = 3072
        qkv_gemm_kernel<<<grid, 128, 0, stream>>>(xb, wqkvT, b_qkv, Qb, Kb, Vt);
    }

    // 3) flash attention (WMMA QK^T + PV, online softmax, double-buffered
    //    async K/V staging)
    {
        dim3 grid(MHA_B * MHA_H * (MHA_S / 128));  // 64 * 16 = 1024
        attn_kernel<<<grid, 256, 0, stream>>>(Qb, Kb, Vt, attnb);
    }

    // 4) output projection (WMMA) with fp32 epilogue
    {
        dim3 grid((MHA_M / 128) * (MHA_E / 64));   // 64 * 16 = 1024
        out_gemm_kernel<<<grid, 128, 0, stream>>>(attnb, woT, b_o, out);
    }
}
